// GAT_22411139350783
// MI455X (gfx1250) — matrix operand order repaired and checked by reading, not compiled, not run on hardware
//
#include <hip/hip_runtime.h>

// ---------------- problem constants (match reference) ----------------
constexpr int   NN    = 100000;         // nodes
constexpr int   EE    = 800000;         // raw edges
constexpr int   ET    = EE + NN;        // edges incl. self loops = 900000
constexpr int   IND   = 128;            // input dim
constexpr int   HID   = 32;             // per-head dim, layer 1
constexpr int   ODIM  = 16;             // per-head dim, layer 2
constexpr int   H     = 8;              // heads
constexpr int   L1O   = H * HID;        // 256
constexpr int   L2O   = H * ODIM;       // 128
constexpr float SLOPE = 0.2f;

typedef __attribute__((ext_vector_type(16))) __bf16          v16bf;
typedef __attribute__((ext_vector_type(16))) unsigned short  v16us;
typedef __attribute__((ext_vector_type(8)))  float           v8f;

// f32 -> bf16 with round-to-nearest-even (bit-level, no __bf16 arithmetic)
__device__ __forceinline__ unsigned short f2bf(float f) {
    unsigned int u = __float_as_uint(f);
    u += 0x7FFFu + ((u >> 16) & 1u);
    return (unsigned short)(u >> 16);
}

// float atomic max via signed-int max / unsigned-int min trick (valid for all
// sign combinations when the cell is initialized to -inf)
__device__ __forceinline__ void atomicMaxFloat(float* addr, float v) {
    if (v >= 0.0f) atomicMax((int*)addr, __float_as_int(v));
    else           atomicMin((unsigned int*)addr, __float_as_uint(v));
}

// ---------------- bf16 WMMA GEMM:  Y[N,OUTW] = X[N,K] * Wrow[OUTW,K]^T ------
// One block = 16 rows. 8 waves; each wave produces TPW 16-column D tiles.
// X tile is staged in LDS as bf16; W is pre-converted bf16 row-major so the
// B fragment per lane is two contiguous 8-element runs (two b128 loads).
template <int K, int OUTW, int TPW>
__global__ __launch_bounds__(256) void gemm_bf16_wmma(
    const float* __restrict__ X,
    const unsigned short* __restrict__ Wbf,
    float* __restrict__ Y)
{
    __shared__ unsigned short lds[16][K + 8];
    const int row0 = blockIdx.x * 16;
    const int tid  = threadIdx.x;

    for (int i = tid; i < 16 * K; i += 256) {
        const int r = i / K, c = i % K;
        lds[r][c] = f2bf(X[(size_t)(row0 + r) * K + c]);
    }
    __syncthreads();

    const int wave = tid >> 5;
    const int lane = tid & 31;
    const int m    = lane & 15;      // A row / D column index
    const int half = lane >> 4;      // lane half selects K sub-range / D row half

    for (int t = 0; t < TPW; ++t) {
        const int ntile = wave * TPW + t;        // 16-col tile index
        const int ncol  = ntile * 16 + m;        // B/D column
        v8f acc = {};
        for (int kk = 0; kk < K; kk += 32) {
            v16us au, bu;
            const unsigned short* a0 = &lds[m][kk + half * 8];
            const unsigned short* b0 = &Wbf[(size_t)ncol * K + kk + half * 8];
#pragma unroll
            for (int j = 0; j < 8; ++j) {        // A: two contiguous 8-runs
                au[j]     = a0[j];
                au[8 + j] = a0[16 + j];
            }
#pragma unroll
            for (int j = 0; j < 8; ++j) {        // B: two contiguous 8-runs
                bu[j]     = b0[j];
                bu[8 + j] = b0[16 + j];
            }
            v16bf a = __builtin_bit_cast(v16bf, au);
            v16bf b = __builtin_bit_cast(v16bf, bu);
            acc = __builtin_amdgcn_wmma_f32_16x16x32_bf16(
                      false, a, false, b, (short)0, acc, false, false);
        }
#pragma unroll
        for (int r = 0; r < 8; ++r)              // D: M = r + 8*half, N = m
            Y[(size_t)(row0 + half * 8 + r) * OUTW + ntile * 16 + m] = acc[r];
    }
}

// ---------------- small helper kernels ----------------
__global__ void convert_weights(const float* __restrict__ w1,
                                const float* __restrict__ w2,
                                unsigned short* __restrict__ w1b,
                                unsigned short* __restrict__ w2b)
{
    const int i = blockIdx.x * 256 + threadIdx.x;
    if (i < L1O * IND) w1b[i] = f2bf(w1[i]);
    if (i < L2O * HID) w2b[i] = f2bf(w2[i]);
}

__global__ void fill_kernel(float* __restrict__ p, float v, int n)
{
    const int i = blockIdx.x * 256 + threadIdx.x;
    if (i < n) p[i] = v;
}

// per-(node,head) attention coefficients: asrc/adst[n*H+h] = <h_row, att>
__global__ void attn_coef_kernel(const float* __restrict__ hmat, int F, int stride,
                                 const float* __restrict__ att_s,
                                 const float* __restrict__ att_d,
                                 float* __restrict__ out_s,
                                 float* __restrict__ out_d)
{
    const int t = blockIdx.x * 256 + threadIdx.x;
    if (t >= NN * H) return;
    const int n = t >> 3, hh = t & 7;
    const float* hp = hmat + (size_t)n * stride + hh * F;
    const float* sp = att_s + hh * F;
    const float* dp = att_d + hh * F;
    float ss = 0.f, sd = 0.f;
    for (int f = 0; f < F; ++f) { const float v = hp[f]; ss += v * sp[f]; sd += v * dp[f]; }
    out_s[t] = ss;
    out_d[t] = sd;
}

__device__ __forceinline__ void edge_sd(const int* __restrict__ ei, int e, int& s, int& d)
{
    if (e < EE) { s = ei[e]; d = ei[EE + e]; }
    else        { s = e - EE; d = s; }
}

// pass 1: e = leaky_relu(asrc[s]+adst[d]); segment max over dst
__global__ void edge_logits_max_kernel(const int* __restrict__ ei,
                                       const float* __restrict__ asb,
                                       const float* __restrict__ adb,
                                       float* __restrict__ ebuf,
                                       float* __restrict__ mbuf)
{
    const int t = blockIdx.x * 256 + threadIdx.x;
    if (t >= ET * H) return;
    const int e = t >> 3, hh = t & 7;
    int s, d; edge_sd(ei, e, s, d);
    float x = asb[s * H + hh] + adb[d * H + hh];
    x = (x > 0.f) ? x : SLOPE * x;
    ebuf[t] = x;
    atomicMaxFloat(&mbuf[d * H + hh], x);
}

// pass 2: p = exp(e - m[dst]); segment sum over dst
__global__ void edge_exp_kernel(const int* __restrict__ ei,
                                float* __restrict__ ebuf,
                                const float* __restrict__ mbuf,
                                float* __restrict__ zbuf)
{
    const int t = blockIdx.x * 256 + threadIdx.x;
    if (t >= ET * H) return;
    const int e = t >> 3, hh = t & 7;
    int s, d; edge_sd(ei, e, s, d); (void)s;
    const float p = __expf(ebuf[t] - mbuf[d * H + hh]);
    ebuf[t] = p;
    atomicAdd(&zbuf[d * H + hh], p);
}

// pass 3: alpha = p/z; scatter (alpha/8)*h[src] into head-mean accumulator
__global__ void edge_scatter_kernel(const int* __restrict__ ei,
                                    const float* __restrict__ ebuf,
                                    const float* __restrict__ zbuf,
                                    const float* __restrict__ hsrc, int F, int stride,
                                    float* __restrict__ alpha_out,
                                    float* __restrict__ acc)
{
    const int t = blockIdx.x * 256 + threadIdx.x;
    if (t >= ET * H) return;
    const int e = t >> 3, hh = t & 7;
    int s, d; edge_sd(ei, e, s, d);
    const float alpha = ebuf[t] / (zbuf[d * H + hh] + 1e-16f);
    alpha_out[t] = alpha;                       // [Etot, H] row-major
    const float c = alpha * 0.125f;             // fold mean over 8 heads
    const float* hp = hsrc + (size_t)s * stride + hh * F;
    float* op = acc + (size_t)d * F;
    for (int f = 0; f < F; ++f) atomicAdd(&op[f], c * hp[f]);
}

__global__ void bias_relu_kernel(float* __restrict__ p, const float* __restrict__ b,
                                 int F, int n)
{
    const int i = blockIdx.x * 256 + threadIdx.x;
    if (i >= n) return;
    const float v = p[i] + b[i % F];
    p[i] = v > 0.f ? v : 0.f;
}

__global__ void out_logsoftmax_kernel(const float* __restrict__ acc,
                                      const float* __restrict__ b2,
                                      float* __restrict__ out)
{
    const int n = blockIdx.x * 256 + threadIdx.x;
    if (n >= NN) return;
    float v[ODIM];
    float mx = -__builtin_inff();
#pragma unroll
    for (int c = 0; c < ODIM; ++c) { v[c] = acc[(size_t)n * ODIM + c] + b2[c]; mx = fmaxf(mx, v[c]); }
    float ssum = 0.f;
#pragma unroll
    for (int c = 0; c < ODIM; ++c) ssum += __expf(v[c] - mx);
    const float l = __logf(ssum);
#pragma unroll
    for (int c = 0; c < ODIM; ++c) out[(size_t)n * ODIM + c] = v[c] - mx - l;
}

__global__ void ei_out_kernel(const int* __restrict__ ei, float* __restrict__ out)
{
    const int i = blockIdx.x * 256 + threadIdx.x;
    if (i >= 2 * ET) return;
    const int part = i / ET, e = i % ET;
    const int v = (e < EE) ? ei[part * EE + e] : (e - EE);
    out[i] = (float)v;
}

// ---------------- host-side launcher ----------------
extern "C" void kernel_launch(void* const* d_in, const int* in_sizes, int n_in,
                              void* d_out, int out_size, void* d_ws, size_t ws_size,
                              hipStream_t stream)
{
    (void)in_sizes; (void)n_in; (void)out_size; (void)ws_size;
    const float* x   = (const float*)d_in[0];
    const int*   ei  = (const int*)  d_in[1];
    const float* W1  = (const float*)d_in[2];
    const float* as1 = (const float*)d_in[3];
    const float* ad1 = (const float*)d_in[4];
    const float* b1  = (const float*)d_in[5];
    const float* W2  = (const float*)d_in[6];
    const float* as2 = (const float*)d_in[7];
    const float* ad2 = (const float*)d_in[8];
    const float* b2  = (const float*)d_in[9];
    float* out = (float*)d_out;

    // workspace carve-up (layer 2 reuses the layer-1 regions)
    char* ws = (char*)d_ws;
    size_t off = 0;
    auto carve = [&](size_t bytes) -> char* {
        char* p = ws + off;
        off = (off + bytes + 255) & ~(size_t)255;
        return p;
    };
    unsigned short* w1b  = (unsigned short*)carve((size_t)L1O * IND * 2);
    unsigned short* w2b  = (unsigned short*)carve((size_t)L2O * HID * 2);
    float* hbuf = (float*)carve((size_t)NN * L1O * 4);   // h1 full, then h2 full
    float* asb  = (float*)carve((size_t)NN * H * 4);
    float* adb  = (float*)carve((size_t)NN * H * 4);
    float* mb   = (float*)carve((size_t)NN * H * 4);
    float* zb   = (float*)carve((size_t)NN * H * 4);
    float* eb   = (float*)carve((size_t)ET * H * 4);
    float* acc1 = (float*)carve((size_t)NN * HID * 4);   // layer-1 mean out -> relu -> layer-2 input
    float* acc2 = (float*)carve((size_t)NN * ODIM * 4);  // layer-2 mean out

    float* out_ls = out;
    float* out_ei = out_ls + (size_t)NN * ODIM;
    float* out_a1 = out_ei + (size_t)2 * ET;
    float* out_a2 = out_a1 + (size_t)ET * H;

    const dim3 blk(256);
    auto g = [](long long n) { return dim3((unsigned)((n + 255) / 256)); };
    const float NEG_INF = -__builtin_inff();

    convert_weights<<<g((long long)L1O * IND), blk, 0, stream>>>(W1, W2, w1b, w2b);

    // ---------------- layer 1 ----------------
    gemm_bf16_wmma<IND, L1O, 2><<<NN / 16, blk, 0, stream>>>(x, w1b, hbuf);
    attn_coef_kernel<<<g((long long)NN * H), blk, 0, stream>>>(hbuf, HID, L1O, as1, ad1, asb, adb);
    fill_kernel<<<g((long long)NN * H), blk, 0, stream>>>(mb, NEG_INF, NN * H);
    fill_kernel<<<g((long long)NN * H), blk, 0, stream>>>(zb, 0.f, NN * H);
    fill_kernel<<<g((long long)NN * HID), blk, 0, stream>>>(acc1, 0.f, NN * HID);
    edge_logits_max_kernel<<<g((long long)ET * H), blk, 0, stream>>>(ei, asb, adb, eb, mb);
    edge_exp_kernel<<<g((long long)ET * H), blk, 0, stream>>>(ei, eb, mb, zb);
    edge_scatter_kernel<<<g((long long)ET * H), blk, 0, stream>>>(ei, eb, zb, hbuf, HID, L1O, out_a1, acc1);
    bias_relu_kernel<<<g((long long)NN * HID), blk, 0, stream>>>(acc1, b1, HID, NN * HID);

    // ---------------- layer 2 ----------------
    gemm_bf16_wmma<HID, L2O, 1><<<NN / 16, blk, 0, stream>>>(acc1, w2b, hbuf);
    attn_coef_kernel<<<g((long long)NN * H), blk, 0, stream>>>(hbuf, ODIM, L2O, as2, ad2, asb, adb);
    fill_kernel<<<g((long long)NN * H), blk, 0, stream>>>(mb, NEG_INF, NN * H);
    fill_kernel<<<g((long long)NN * H), blk, 0, stream>>>(zb, 0.f, NN * H);
    fill_kernel<<<g((long long)NN * ODIM), blk, 0, stream>>>(acc2, 0.f, NN * ODIM);
    edge_logits_max_kernel<<<g((long long)ET * H), blk, 0, stream>>>(ei, asb, adb, eb, mb);
    edge_exp_kernel<<<g((long long)ET * H), blk, 0, stream>>>(ei, eb, mb, zb);
    edge_scatter_kernel<<<g((long long)ET * H), blk, 0, stream>>>(ei, eb, zb, hbuf, ODIM, L2O, out_a2, acc2);

    // ---------------- outputs ----------------
    out_logsoftmax_kernel<<<g((long long)NN), blk, 0, stream>>>(acc2, b2, out_ls);
    ei_out_kernel<<<g((long long)2 * ET), blk, 0, stream>>>(ei, out_ei);
}
